// KimiDeltaAttention_52398601011566
// MI455X (gfx1250) — compile-verified, matrix operand order checked by reading, not run on hardware
//
#include <hip/hip_runtime.h>

// ---------------- problem constants ----------------
#define NTOK 4096
#define HID  2048
#define NH   16
#define HD   128
#define PROJ 2048   // NH*HD
#define KCONV 4

typedef __attribute__((ext_vector_type(16))) __bf16 v16bf;
typedef __attribute__((ext_vector_type(8)))  __bf16 v8bf;
typedef __attribute__((ext_vector_type(8)))  float  v8f;

// ---------------- fp32 -> bf16 convert ----------------
__global__ void cvt_f32_bf16(const float* __restrict__ in, __bf16* __restrict__ out, int count) {
    int i = blockIdx.x * blockDim.x + threadIdx.x;
    if (i < count) out[i] = (__bf16)in[i];
}

// ---------------- bf16 WMMA GEMM: C(f32 MxN) = A(bf16 MxK) * B(bf16 KxN) ----------------
// block tile 64x128, 256 threads = 8 waves in 2x4, each wave 2x2 16x16 frags.
// BK=64: two WMMA K-steps per LDS stage (8 wmma per wave per barrier).
// A tile staged with CDNA5 async copies (global_load_async_to_lds_b128, ASYNCcnt);
// B tile staged transposed (software) so fragments load as contiguous 16B groups.
#define BM 64
#define BN 128
#define BK 64
#define LPAD 16   // pad elements; keeps row stride (BK+LPAD)*2 = 160B, multiple of 16B
__global__ __launch_bounds__(256) void gemm_bf16_wmma(const __bf16* __restrict__ A,
                                                      const __bf16* __restrict__ B,
                                                      float* __restrict__ C,
                                                      int M, int N, int K) {
    __shared__ __bf16 As[BM][BK + LPAD];    // M x K, row-major
    __shared__ __bf16 BsT[BN][BK + LPAD];   // N x K (transposed during staging)
    const int tid  = threadIdx.x;
    const int lane = tid & 31;
    const int wave = tid >> 5;
    const int wm   = (wave >> 2) * 32;   // 0 or 32
    const int wn   = (wave & 3) * 32;    // 0,32,64,96
    const int bm0  = blockIdx.y * BM;
    const int bn0  = blockIdx.x * BN;

    const int row16 = lane & 15;
    const int khalf = (lane >> 4) ? 8 : 0;   // K sub-block per ISA 16-bit operand layout
    const int mhalf = (lane >> 4) ? 8 : 0;   // C/D row half

    v8f zero = {};
    v8f acc[2][2];
    acc[0][0] = zero; acc[0][1] = zero; acc[1][0] = zero; acc[1][1] = zero;

    for (int k0 = 0; k0 < K; k0 += BK) {
        // ---- prefetch next B tile (global_prefetch_b8) ----
        if (k0 + BK < K && tid < 128) {
            int r = tid & 63;
            __builtin_prefetch(&B[(size_t)(k0 + BK + r) * N + bn0 + ((tid & 64) ? 64 : 0)], 0, 3);
        }
        // ---- stage A tile (64x64) via async global->LDS copies (no VGPR round-trip) ----
        #pragma unroll
        for (int u = 0; u < 2; u++) {
            int q = tid + 256 * u;             // 512 chunks of 8 bf16 (16B)
            int r = q >> 3;                    // row 0..63
            int c = (q & 7) << 3;              // col 0,8,...,56
            const __bf16* gp = A + (size_t)(bm0 + r) * K + k0 + c;
            unsigned loff = (unsigned)(size_t)&As[r][c];   // flat->LDS: addr[31:0]
            asm volatile("global_load_async_to_lds_b128 %0, %1, off"
                         :: "v"(loff), "v"(gp) : "memory");
        }
        // ---- stage B tile (64x128) transposed into BsT ----
        #pragma unroll
        for (int u = 0; u < 4; u++) {
            int q  = tid + 256 * u;        // 0..1023 chunks of 8 elems
            int r  = q >> 4;               // K row 0..63
            int c8 = (q & 15) << 3;        // N col 0,8,...,120
            union { uint4 u4; __bf16 h[8]; } tmp;
            tmp.u4 = *(const uint4*)(B + (size_t)(k0 + r) * N + bn0 + c8);
            #pragma unroll
            for (int j = 0; j < 8; j++) BsT[c8 + j][r] = tmp.h[j];
        }
        // wait for async A-tile copies, then workgroup barrier
        asm volatile("s_wait_asynccnt 0" ::: "memory");
        __syncthreads();

        // ---- two WMMA K-steps over the staged 64-deep tile ----
        #pragma unroll
        for (int kk = 0; kk < BK; kk += 32) {
            v16bf afrag[2], bfrag[2];
            #pragma unroll
            for (int i = 0; i < 2; i++) {
                int m = wm + i * 16 + row16;
                v8bf lo = *(const v8bf*)&As[m][kk + khalf];
                v8bf hi = *(const v8bf*)&As[m][kk + khalf + 16];
                afrag[i] = __builtin_shufflevector(lo, hi, 0,1,2,3,4,5,6,7,8,9,10,11,12,13,14,15);
            }
            #pragma unroll
            for (int i = 0; i < 2; i++) {
                int n = wn + i * 16 + row16;
                v8bf lo = *(const v8bf*)&BsT[n][kk + khalf];
                v8bf hi = *(const v8bf*)&BsT[n][kk + khalf + 16];
                bfrag[i] = __builtin_shufflevector(lo, hi, 0,1,2,3,4,5,6,7,8,9,10,11,12,13,14,15);
            }
            #pragma unroll
            for (int i = 0; i < 2; i++)
                #pragma unroll
                for (int j = 0; j < 2; j++)
                    acc[i][j] = __builtin_amdgcn_wmma_f32_16x16x32_bf16(
                        false, afrag[i], false, bfrag[j], (short)0, acc[i][j], false, false);
        }
        __syncthreads();
    }

    #pragma unroll
    for (int i = 0; i < 2; i++)
        #pragma unroll
        for (int j = 0; j < 2; j++) {
            int n = bn0 + wn + j * 16 + row16;
            #pragma unroll
            for (int r = 0; r < 8; r++) {
                int m = bm0 + wm + i * 16 + mhalf + r;
                C[(size_t)m * N + n] = acc[i][j][r];
            }
        }
}

// ---------------- causal depthwise conv (K=4) + SiLU ----------------
__global__ void conv_silu(const float* __restrict__ x, const float* __restrict__ w,
                          float* __restrict__ y, int total) {
    int i = blockIdx.x * blockDim.x + threadIdx.x;
    if (i >= total) return;
    int t = i >> 11;            // / PROJ
    int c = i & (PROJ - 1);
    float acc = 0.f;
    #pragma unroll
    for (int j = 0; j < KCONV; j++) {
        int tt = t - (KCONV - 1) + j;
        if (tt >= 0) acc += x[(size_t)tt * PROJ + c] * w[c * KCONV + j];
    }
    y[i] = acc / (1.f + __expf(-acc));   // silu
}

// ---------------- beta = sigmoid(h @ Wb) ----------------
__global__ void beta_kernel(const float* __restrict__ h, const float* __restrict__ Wb,
                            float* __restrict__ beta) {
    int i = blockIdx.x * blockDim.x + threadIdx.x;
    if (i >= NTOK * NH) return;
    int t = i >> 4, j = i & 15;
    const float* hp = h + (size_t)t * HID;
    float s = 0.f;
    for (int c = 0; c < HID; c++) s += hp[c] * Wb[c * NH + j];
    beta[i] = 1.f / (1.f + __expf(-s));
}

// ---------------- KDA gate: g1 = -exp(A_log[h]) * softplus(g1lin + dt_bias) (in place) ----------------
__global__ void g1_act(float* __restrict__ g, const float* __restrict__ dt_bias,
                       const float* __restrict__ A_log, int total) {
    int i = blockIdx.x * blockDim.x + threadIdx.x;
    if (i >= total) return;
    int c = i & (PROJ - 1);
    int h = c >> 7;                     // / HD
    float x = g[i] + dt_bias[c];
    float sp = (x > 20.f) ? x : log1pf(__expf(x));
    g[i] = -__expf(A_log[h]) * sp;
}

// ---------------- in-place l2norm over D=128, one wave32 per row ----------------
__global__ void l2norm_rows(float* __restrict__ x, int rows, float scale) {
    int lane = threadIdx.x & 31;
    int row  = blockIdx.x * (blockDim.x >> 5) + (threadIdx.x >> 5);
    if (row >= rows) return;
    float* p = x + (size_t)row * HD;
    float v0 = p[lane], v1 = p[lane + 32], v2 = p[lane + 64], v3 = p[lane + 96];
    float ss = v0 * v0 + v1 * v1 + v2 * v2 + v3 * v3;
    #pragma unroll
    for (int m = 16; m >= 1; m >>= 1) ss += __shfl_xor(ss, m, 32);
    float r = rsqrtf(ss + 1e-6f) * scale;
    p[lane] = v0 * r; p[lane + 32] = v1 * r; p[lane + 64] = v2 * r; p[lane + 96] = v3 * r;
}

// ---------------- sequential delta-rule scan ----------------
// grid (2, NH): blockIdx.y = head, blockIdx.x = 64-wide v-column chunk.
// 256 threads: v_local = tid&63, kg = tid>>6 owns 32 k-rows of the state column.
__global__ __launch_bounds__(256) void kda_scan(const float* __restrict__ q,
                                                const float* __restrict__ k,
                                                const float* __restrict__ v,
                                                const float* __restrict__ g,
                                                const float* __restrict__ beta,
                                                float* __restrict__ o) {
    const int tid   = threadIdx.x;
    const int vloc  = tid & 63;
    const int kg    = tid >> 6;         // 0..3
    const int kbase = kg * 32;
    const int head  = blockIdx.y;
    const int vbase = blockIdx.x * 64;
    const int cbase = head * HD;

    __shared__ float kk[HD], qq[HD], gg[HD], vv[64];
    __shared__ float pmem[4][64], po[4][64], delta_s[64];
    __shared__ float bsh;

    float S[32];
    #pragma unroll
    for (int i = 0; i < 32; i++) S[i] = 0.f;

    for (int t = 0; t < NTOK; t++) {
        const size_t rowb = (size_t)t * PROJ + cbase;
        if (tid < HD) {
            kk[tid] = k[rowb + tid];
            qq[tid] = q[rowb + tid];
            gg[tid] = __expf(g[rowb + tid]);
        }
        if (tid < 64) vv[tid] = v[rowb + vbase + tid];
        if (tid == 0) bsh = beta[t * NH + head];
        __syncthreads();

        // decay state, then mem[v] = sum_k kk[k] * S[k][v]
        float p = 0.f;
        #pragma unroll
        for (int i = 0; i < 32; i++) {
            S[i] *= gg[kbase + i];
            p += kk[kbase + i] * S[i];
        }
        pmem[kg][vloc] = p;
        __syncthreads();

        if (kg == 0)
            delta_s[vloc] = (vv[vloc] - (pmem[0][vloc] + pmem[1][vloc] + pmem[2][vloc] + pmem[3][vloc])) * bsh;
        __syncthreads();

        const float d = delta_s[vloc];
        float p2 = 0.f;
        #pragma unroll
        for (int i = 0; i < 32; i++) {
            S[i] += kk[kbase + i] * d;
            p2 += qq[kbase + i] * S[i];
        }
        po[kg][vloc] = p2;
        __syncthreads();

        if (kg == 0)
            o[rowb + vbase + vloc] = po[0][vloc] + po[1][vloc] + po[2][vloc] + po[3][vloc];
        // next iteration's first barrier protects pmem/po reuse
    }
}

// ---------------- gated RMSNorm + sigmoid(g2) gate -> bf16, one wave per (t,h) row ----------------
__global__ void gated_norm(const float* __restrict__ o, const float* __restrict__ g2,
                           const float* __restrict__ w, __bf16* __restrict__ out, int rows) {
    int lane = threadIdx.x & 31;
    int row  = blockIdx.x * (blockDim.x >> 5) + (threadIdx.x >> 5);
    if (row >= rows) return;
    const size_t base = (size_t)row * HD;
    float v0 = o[base + lane], v1 = o[base + lane + 32], v2 = o[base + lane + 64], v3 = o[base + lane + 96];
    float ss = v0 * v0 + v1 * v1 + v2 * v2 + v3 * v3;
    #pragma unroll
    for (int m = 16; m >= 1; m >>= 1) ss += __shfl_xor(ss, m, 32);
    float r = rsqrtf(ss * (1.0f / HD) + 1e-5f);
    #pragma unroll
    for (int e = 0; e < 4; e++) {
        int d = lane + e * 32;
        float val = (e == 0 ? v0 : e == 1 ? v1 : e == 2 ? v2 : v3);
        float gate = 1.f / (1.f + __expf(-g2[base + d]));
        out[base + d] = (__bf16)(val * r * w[d] * gate);
    }
}

// ---------------- host side ----------------
static inline size_t align256(size_t x) { return (x + 255) & ~(size_t)255; }

extern "C" void kernel_launch(void* const* d_in, const int* in_sizes, int n_in,
                              void* d_out, int out_size, void* d_ws, size_t ws_size,
                              hipStream_t stream) {
    const float* hs      = (const float*)d_in[0];
    // d_in[1] positions: arange, unused for prefill
    const float* Wq      = (const float*)d_in[2];
    const float* Wk      = (const float*)d_in[3];
    const float* Wv      = (const float*)d_in[4];
    const float* Wfa     = (const float*)d_in[5];
    const float* Wfb     = (const float*)d_in[6];
    const float* dt_bias = (const float*)d_in[7];
    const float* Wb      = (const float*)d_in[8];
    const float* conv_q  = (const float*)d_in[9];
    const float* conv_k  = (const float*)d_in[10];
    const float* conv_v  = (const float*)d_in[11];
    const float* A_log   = (const float*)d_in[12];
    const float* Wga     = (const float*)d_in[13];
    const float* Wgb     = (const float*)d_in[14];
    const float* o_norm_w= (const float*)d_in[15];
    const float* Wo      = (const float*)d_in[16];
    float* out = (float*)d_out;

    // -------- workspace carve --------
    char* p = (char*)d_ws;
    auto carve = [&](size_t bytes) { char* r = p; p += align256(bytes); return (void*)r; };
    __bf16* hsb   = (__bf16*)carve((size_t)NTOK * HID * 2);
    __bf16* wq_b  = (__bf16*)carve((size_t)HID * PROJ * 2);
    __bf16* wk_b  = (__bf16*)carve((size_t)HID * PROJ * 2);
    __bf16* wv_b  = (__bf16*)carve((size_t)HID * PROJ * 2);
    __bf16* wo_b  = (__bf16*)carve((size_t)PROJ * HID * 2);
    __bf16* wfa_b = (__bf16*)carve((size_t)HID * HD * 2);
    __bf16* wfb_b = (__bf16*)carve((size_t)HD * PROJ * 2);
    __bf16* wga_b = (__bf16*)carve((size_t)HID * HD * 2);
    __bf16* wgb_b = (__bf16*)carve((size_t)HD * PROJ * 2);
    float*  qlin  = (float*)carve((size_t)NTOK * PROJ * 4);   // later reused for g1
    float*  klin  = (float*)carve((size_t)NTOK * PROJ * 4);   // later reused for g2
    float*  vlin  = (float*)carve((size_t)NTOK * PROJ * 4);   // later reused for o
    float*  qc    = (float*)carve((size_t)NTOK * PROJ * 4);   // later reused for ob (bf16)
    float*  kc    = (float*)carve((size_t)NTOK * PROJ * 4);
    float*  vc    = (float*)carve((size_t)NTOK * PROJ * 4);
    float*  fa32  = (float*)carve((size_t)NTOK * HD * 4);
    float*  ga32  = (float*)carve((size_t)NTOK * HD * 4);
    __bf16* fab   = (__bf16*)carve((size_t)NTOK * HD * 2);
    __bf16* gab   = (__bf16*)carve((size_t)NTOK * HD * 2);
    float*  betab = (float*)carve((size_t)NTOK * NH * 4);
    (void)ws_size; (void)in_sizes; (void)n_in; (void)out_size;

    const int THR = 256;
    auto nblk = [&](size_t c) { return (unsigned)((c + THR - 1) / THR); };

    // 1) bf16 conversions
    cvt_f32_bf16<<<nblk((size_t)NTOK*HID), THR, 0, stream>>>(hs,  hsb,  NTOK*HID);
    cvt_f32_bf16<<<nblk((size_t)HID*PROJ), THR, 0, stream>>>(Wq,  wq_b, HID*PROJ);
    cvt_f32_bf16<<<nblk((size_t)HID*PROJ), THR, 0, stream>>>(Wk,  wk_b, HID*PROJ);
    cvt_f32_bf16<<<nblk((size_t)HID*PROJ), THR, 0, stream>>>(Wv,  wv_b, HID*PROJ);
    cvt_f32_bf16<<<nblk((size_t)PROJ*HID), THR, 0, stream>>>(Wo,  wo_b, PROJ*HID);
    cvt_f32_bf16<<<nblk((size_t)HID*HD),   THR, 0, stream>>>(Wfa, wfa_b, HID*HD);
    cvt_f32_bf16<<<nblk((size_t)HD*PROJ),  THR, 0, stream>>>(Wfb, wfb_b, HD*PROJ);
    cvt_f32_bf16<<<nblk((size_t)HID*HD),   THR, 0, stream>>>(Wga, wga_b, HID*HD);
    cvt_f32_bf16<<<nblk((size_t)HD*PROJ),  THR, 0, stream>>>(Wgb, wgb_b, HD*PROJ);

    // 2) projection GEMMs (WMMA)
    dim3 gP(PROJ / BN, NTOK / BM), b256(256);
    gemm_bf16_wmma<<<gP, b256, 0, stream>>>(hsb, wq_b, qlin, NTOK, PROJ, HID);
    gemm_bf16_wmma<<<gP, b256, 0, stream>>>(hsb, wk_b, klin, NTOK, PROJ, HID);
    gemm_bf16_wmma<<<gP, b256, 0, stream>>>(hsb, wv_b, vlin, NTOK, PROJ, HID);
    dim3 gD(HD / BN, NTOK / BM);
    gemm_bf16_wmma<<<gD, b256, 0, stream>>>(hsb, wfa_b, fa32, NTOK, HD, HID);
    gemm_bf16_wmma<<<gD, b256, 0, stream>>>(hsb, wga_b, ga32, NTOK, HD, HID);

    // 3) conv + silu (frees qlin/klin/vlin afterwards)
    conv_silu<<<nblk((size_t)NTOK*PROJ), THR, 0, stream>>>(qlin, conv_q, qc, NTOK*PROJ);
    conv_silu<<<nblk((size_t)NTOK*PROJ), THR, 0, stream>>>(klin, conv_k, kc, NTOK*PROJ);
    conv_silu<<<nblk((size_t)NTOK*PROJ), THR, 0, stream>>>(vlin, conv_v, vc, NTOK*PROJ);

    // 4) low-rank gates: g1 = act(fa @ Wfb) into qlin ; g2 = ga @ Wgb into klin
    cvt_f32_bf16<<<nblk((size_t)NTOK*HD), THR, 0, stream>>>(fa32, fab, NTOK*HD);
    cvt_f32_bf16<<<nblk((size_t)NTOK*HD), THR, 0, stream>>>(ga32, gab, NTOK*HD);
    gemm_bf16_wmma<<<gP, b256, 0, stream>>>(fab, wfb_b, qlin, NTOK, PROJ, HD);
    gemm_bf16_wmma<<<gP, b256, 0, stream>>>(gab, wgb_b, klin, NTOK, PROJ, HD);
    g1_act<<<nblk((size_t)NTOK*PROJ), THR, 0, stream>>>(qlin, dt_bias, A_log, NTOK*PROJ);

    // 5) beta
    beta_kernel<<<nblk((size_t)NTOK*NH), THR, 0, stream>>>(hs, Wb, betab);

    // 6) l2norm q (scaled by D^-0.5) and k, in place
    const int rows = NTOK * NH;
    l2norm_rows<<<(rows + 7) / 8, THR, 0, stream>>>(qc, rows, rsqrtf((float)HD));
    l2norm_rows<<<(rows + 7) / 8, THR, 0, stream>>>(kc, rows, 1.0f);

    // 7) sequential delta-rule scan -> o in vlin
    dim3 gS(2, NH);
    kda_scan<<<gS, b256, 0, stream>>>(qc, kc, vc, qlin /*g1*/, betab, vlin /*o*/);

    // 8) gated RMSNorm -> bf16 (reuse qc region)
    __bf16* ob = (__bf16*)qc;
    gated_norm<<<(rows + 7) / 8, THR, 0, stream>>>(vlin /*o*/, klin /*g2*/, o_norm_w, ob, rows);

    // 9) output projection straight to d_out (fp32)
    dim3 gO(HID / BN, NTOK / BM);
    gemm_bf16_wmma<<<gO, b256, 0, stream>>>(ob, wo_b, out, NTOK, HID, PROJ);
}
